// AttentionLayer_31619549233446
// MI455X (gfx1250) — matrix operand order repaired and checked by reading, not compiled, not run on hardware
//
#include <hip/hip_runtime.h>
#include <math.h>

#define Bdim 32
#define Tdim 2048
#define Ddim 512
#define Udim 512

typedef __attribute__((ext_vector_type(16))) _Float16    v16h;
typedef __attribute__((ext_vector_type(8)))  float       v8f;
typedef __attribute__((ext_vector_type(4))) unsigned int uivec4;
typedef __attribute__((ext_vector_type(4)))  float       fvec4;

struct Pack32 { uivec4 lo, hi; };   // 32 bytes == one v16h fragment

static __device__ __forceinline__ unsigned int f2h2(float a, float b) {
  _Float16 ha = (_Float16)a;
  _Float16 hb = (_Float16)b;
  return (unsigned int)__builtin_bit_cast(unsigned short, ha) |
         ((unsigned int)__builtin_bit_cast(unsigned short, hb) << 16);
}

// ---------------------------------------------------------------------------
// 1) Wk [D,U] f32  ->  WkT [U,D] f16   (B-matrix rows contiguous per column n)
// ---------------------------------------------------------------------------
__global__ void wkt_pack_kernel(const float* __restrict__ Wk,
                                _Float16* __restrict__ WkT) {
  int idx = blockIdx.x * blockDim.x + threadIdx.x;   // 0 .. 262143
  int u = idx >> 9;
  int d = idx & 511;
  WkT[idx] = (_Float16)Wk[d * Udim + u];
}

// ---------------------------------------------------------------------------
// 2) uh[b,u] = hidden[b,:] @ Uk[:,u] + Ub[u] + Wb[u]   (Wb folded in)
// ---------------------------------------------------------------------------
__global__ void uh_kernel(const float* __restrict__ hidden,
                          const float* __restrict__ Uk,
                          const float* __restrict__ Ub,
                          const float* __restrict__ Wb,
                          float* __restrict__ uh) {
  int b = blockIdx.x;
  int u = threadIdx.x;                       // 0..255, handles u and u+256
  float acc0 = Ub[u] + Wb[u];
  float acc1 = Ub[u + 256] + Wb[u + 256];
  for (int k = 0; k < 512; ++k) {
    float h = hidden[b * 512 + k];
    acc0 += h * Uk[k * Udim + u];
    acc1 += h * Uk[k * Udim + u + 256];
  }
  uh[b * Udim + u] = acc0;
  uh[b * Udim + u + 256] = acc1;
}

// ---------------------------------------------------------------------------
// Async-stage one 16-row (16 KB) B tile of WkT into LDS, XOR-swizzled in 16B
// blocks. 256 threads x 4 blocks; tracked by ASYNCcnt.
// ---------------------------------------------------------------------------
static __device__ __forceinline__ void stage_b_async(
    const _Float16* __restrict__ WkT, int n0,
    unsigned short* bufbase, int tid) {
  #pragma unroll
  for (int i = 0; i < 4; ++i) {
    int g = tid + i * 256;                 // 16B block id: 0..1023
    int row = g >> 6;                      // 0..15
    int blk = g & 63;                      // block within row
    const _Float16* gp = WkT + (size_t)(n0 + row) * Ddim + blk * 8;
    unsigned int laddr =
        (unsigned int)(size_t)(&bufbase[row * 512 + ((blk ^ row) << 3)]);
    asm volatile("global_load_async_to_lds_b128 %0, %1, off"
                 :: "v"(laddr), "v"(gp)
                 : "memory");
  }
}

// ---------------------------------------------------------------------------
// 3) Fused WMMA kernel: logits[b,t] = sum_u tanh( (F@Wk)[t,u] + uh[b,u] )*Vk[u]
//    256 threads (8 waves), 128 T-rows per WG. A in registers (direct from
//    global, f32->f16), B shared across waves via double-buffered async LDS.
//    B fragments are double-buffered in registers (p/q) so each WMMA waits
//    only s_wait_dscnt 0x2, hiding LDS latency behind the previous WMMA.
// ---------------------------------------------------------------------------
__global__ __launch_bounds__(256) void attn_logits_kernel(
    const float* __restrict__ features, const _Float16* __restrict__ WkT,
    const float* __restrict__ uh, const float* __restrict__ Vk,
    const float* __restrict__ Vb, float* __restrict__ logits) {
  __shared__ __align__(16) unsigned short smem[2 * 16 * 512];   // 2 x 16KB
  const int ttile = blockIdx.x;            // 0..15
  const int b     = blockIdx.y;            // 0..31
  const int tid   = threadIdx.x;           // 0..255
  const int wave  = tid >> 5;              // 0..7
  const int lane  = tid & 31;
  const int hilo  = lane >> 4;             // fragment half (ISA 16-bit layout)
  const int ln    = lane & 15;
  const int rowbase = ttile * 128;
  const int trow  = rowbase + wave * 16 + ln;   // this lane's feature row

  // ---- A: 16 rows x 512 K of features -> 16 f16 fragments in registers ----
  // A 16x32 layout: lane half holds K {0..7,16..23} vs {8..15,24..31}.
  v16h a[16];
  {
    const float* fr = features + ((size_t)b * Tdim + trow) * Ddim + hilo * 8;
    #pragma unroll
    for (int ks = 0; ks < 16; ++ks) {
      const float* p0 = fr + ks * 32;      // floats k0 + hilo*8
      const float* p1 = p0 + 16;           // floats k0 + 16 + hilo*8
      fvec4 f0 = *reinterpret_cast<const fvec4*>(p0);
      fvec4 f1 = *reinterpret_cast<const fvec4*>(p0 + 4);
      fvec4 f2 = *reinterpret_cast<const fvec4*>(p1);
      fvec4 f3 = *reinterpret_cast<const fvec4*>(p1 + 4);
      Pack32 pk;
      uivec4 lo = {f2h2(f0.x, f0.y), f2h2(f0.z, f0.w),
                   f2h2(f1.x, f1.y), f2h2(f1.z, f1.w)};
      uivec4 hi = {f2h2(f2.x, f2.y), f2h2(f2.z, f2.w),
                   f2h2(f3.x, f3.y), f2h2(f3.z, f3.w)};
      pk.lo = lo;
      pk.hi = hi;
      a[ks] = __builtin_bit_cast(v16h, pk);
    }
  }

  // Prologue: stage B tile 0 into buffer 0.
  stage_b_async(WkT, 0, smem, tid);
  asm volatile("s_wait_asynccnt 0" ::: "memory");
  __syncthreads();

  float rowacc[8];
  #pragma unroll
  for (int v = 0; v < 8; ++v) rowacc[v] = 0.f;

  for (int nt = 0; nt < 32; ++nt) {
    unsigned short* bcur = smem + (nt & 1) * 8192;
    if (nt + 1 < 32) {   // prefetch next B tile into the other buffer
      stage_b_async(WkT, (nt + 1) * 16, smem + ((nt + 1) & 1) * 8192, tid);
    }
    const int n0 = nt * 16;
    const float uh_n = uh[b * Udim + n0 + ln];   // this lane's column bias
    const float vk_n = Vk[n0 + ln];              // this lane's column V weight

    // B fragment for column n = n0+ln: halves k0 + hilo*8 .. +7, k0 + 16 + hilo*8 .. +7
    const unsigned short* brow = bcur + ln * 512;
    const int blkbase = hilo;
    auto ldb = [&](int ks, uivec4& o0, uivec4& o1) {
      int blk0 = (ks << 2) + blkbase;            // 16B block within row
      o0 = *reinterpret_cast<const uivec4*>(&brow[(blk0 ^ ln) << 3]);
      o1 = *reinterpret_cast<const uivec4*>(&brow[((blk0 + 2) ^ ln) << 3]);
    };

    v8f c = {};
    uivec4 p0, p1, q0, q1;                       // two in-flight fragments
    ldb(0, p0, p1);
    ldb(1, q0, q1);
    #pragma unroll
    for (int ks = 0; ks < 16; ks += 2) {
      {
        Pack32 pb{p0, p1};
        v16h bv = __builtin_bit_cast(v16h, pb);
        c = __builtin_amdgcn_wmma_f32_16x16x32_f16(false, a[ks], false, bv,
                                                   (short)0, c, false, false);
      }
      if (ks + 2 < 16) ldb(ks + 2, p0, p1);      // refill p after consuming it
      {
        Pack32 pb{q0, q1};
        v16h bv = __builtin_bit_cast(v16h, pb);
        c = __builtin_amdgcn_wmma_f32_16x16x32_f16(false, a[ks + 1], false, bv,
                                                   (short)0, c, false, false);
      }
      if (ks + 3 < 16) ldb(ks + 3, q0, q1);      // refill q after consuming it
    }
    // C/D layout: VGPR v, lane -> (M = v + 8*hilo, N = ln). Fuse tanh + Vk.
    #pragma unroll
    for (int v = 0; v < 8; ++v) {
      rowacc[v] += tanhf(c[v] + uh_n) * vk_n;
    }
    // Own async writes done + everyone finished reading before buffer reuse.
    asm volatile("s_wait_asynccnt 0" ::: "memory");
    __syncthreads();
  }

  // Sum partial column contributions across the 16 lanes of each half.
  #pragma unroll
  for (int v = 0; v < 8; ++v) {
    float x = rowacc[v];
    x += __shfl_xor(x, 1, 32);
    x += __shfl_xor(x, 2, 32);
    x += __shfl_xor(x, 4, 32);
    x += __shfl_xor(x, 8, 32);
    rowacc[v] = x;
  }
  if (ln == 0) {                           // lanes 0 (rows 0-7) and 16 (8-15)
    const float vb = Vb[0];
    #pragma unroll
    for (int v = 0; v < 8; ++v) {
      int t = rowbase + wave * 16 + hilo * 8 + v;
      logits[b * Tdim + t] = rowacc[v] + vb;
    }
  }
}

// ---------------------------------------------------------------------------
// 4) softmax over T per batch -> attention_weights output
// ---------------------------------------------------------------------------
__global__ void softmax_kernel(const float* __restrict__ logits,
                               float* __restrict__ attn) {
  __shared__ float red[256];
  const int b = blockIdx.x;
  const int tid = threadIdx.x;
  float v[8];
  float m = -3.4e38f;
  #pragma unroll
  for (int i = 0; i < 8; ++i) {
    v[i] = logits[b * Tdim + tid + i * 256];
    m = fmaxf(m, v[i]);
  }
  red[tid] = m;
  __syncthreads();
  for (int s = 128; s > 0; s >>= 1) {
    if (tid < s) red[tid] = fmaxf(red[tid], red[tid + s]);
    __syncthreads();
  }
  m = red[0];
  __syncthreads();
  float sum = 0.f;
  #pragma unroll
  for (int i = 0; i < 8; ++i) { v[i] = __expf(v[i] - m); sum += v[i]; }
  red[tid] = sum;
  __syncthreads();
  for (int s = 128; s > 0; s >>= 1) {
    if (tid < s) red[tid] += red[tid + s];
    __syncthreads();
  }
  const float inv = 1.0f / red[0];
  #pragma unroll
  for (int i = 0; i < 8; ++i)
    attn[b * Tdim + tid + i * 256] = v[i] * inv;
}

// ---------------------------------------------------------------------------
// 5) context[b,d] = sum_t attn[b,t] * features[b,t,d]   (second features pass)
// ---------------------------------------------------------------------------
__global__ void context_kernel(const float* __restrict__ attn,
                               const float* __restrict__ features,
                               float* __restrict__ ctx) {
  const int b = blockIdx.x >> 1;
  const int d = ((blockIdx.x & 1) << 8) + threadIdx.x;   // 0..511
  const float* ab = attn + b * Tdim;
  const float* fb = features + (size_t)b * Tdim * Ddim + d;
  float acc = 0.f;
  for (int t = 0; t < Tdim; ++t) {
    acc += ab[t] * fb[(size_t)t * Ddim];
  }
  ctx[b * Ddim + d] = acc;
}

// ---------------------------------------------------------------------------
extern "C" void kernel_launch(void* const* d_in, const int* in_sizes, int n_in,
                              void* d_out, int out_size, void* d_ws, size_t ws_size,
                              hipStream_t stream) {
  (void)in_sizes; (void)n_in; (void)out_size; (void)ws_size;
  const float* features = (const float*)d_in[0];
  const float* hidden   = (const float*)d_in[1];
  const float* Wk       = (const float*)d_in[2];
  const float* Wb       = (const float*)d_in[3];
  const float* Uk       = (const float*)d_in[4];
  const float* Ub       = (const float*)d_in[5];
  const float* Vk       = (const float*)d_in[6];
  const float* Vb       = (const float*)d_in[7];

  float* out_ctx  = (float*)d_out;                 // [B, D]
  float* out_attn = out_ctx + Bdim * Ddim;         // [B, T, 1]

  char* ws = (char*)d_ws;
  _Float16* WkT = (_Float16*)ws;                            // 512 KB
  float*    uh  = (float*)(ws + 512 * 1024);                // 64 KB
  float*    lg  = (float*)(ws + 512 * 1024 + 64 * 1024);    // 256 KB

  wkt_pack_kernel<<<dim3((Ddim * Udim) / 256), 256, 0, stream>>>(Wk, WkT);
  uh_kernel<<<dim3(Bdim), 256, 0, stream>>>(hidden, Uk, Ub, Wb, uh);
  attn_logits_kernel<<<dim3(Tdim / 128, Bdim), 256, 0, stream>>>(
      features, WkT, uh, Vk, Vb, lg);
  softmax_kernel<<<dim3(Bdim), 256, 0, stream>>>(lg, out_attn);
  context_kernel<<<dim3(Bdim * 2), 256, 0, stream>>>(out_attn, features, out_ctx);
}
